// Year_basis_1949915153223
// MI455X (gfx1250) — matrix operand order repaired and checked by reading, not compiled, not run on hardware
//
#include <hip/hip_runtime.h>
#include <math.h>

typedef __attribute__((ext_vector_type(16))) _Float16 v16h;
typedef __attribute__((ext_vector_type(8)))  _Float16 v8h;
typedef __attribute__((ext_vector_type(4)))  _Float16 v4h;
typedef __attribute__((ext_vector_type(8)))  float    v8f;
typedef __attribute__((ext_vector_type(4)))  float    v4f;

#define TB 336
#define TF 96
#define KDIM 21
#define PERIOD 366
#define HID 168
#define ROWS 86016           /* 4096*21 */
#define KPAD 352             /* 11*32   */
#define NT 11                /* N tiles: 176/16 (H padded) */
#define KK 11                /* K steps: 352/32 */

__global__ void __launch_bounds__(128)
nbeats_gate_kernel(const float* __restrict__ theta,
                   const int*   __restrict__ x,
                   const float* __restrict__ basis,
                   const float* __restrict__ W1,
                   const float* __restrict__ B1v,
                   const float* __restrict__ W2,
                   const float* __restrict__ B2v,
                   const float* __restrict__ weight,
                   const float* __restrict__ noise,
                   float* __restrict__ out,
                   float* __restrict__ gacc)
{
  __shared__ _Float16 s_inp[16][KPAD];   // inp tile, f16 (A matrix)
  __shared__ float    s_bb[16][TB];      // centered bb, reused for output
  __shared__ float    s_bbmean[16];
  __shared__ float    s_sample[16];
  __shared__ float    s_hsum[16];
  __shared__ int      s_idx[16];
  __shared__ float    s_red[48];

  const int tid  = threadIdx.x;
  const int w    = tid >> 5;       // wave id 0..3
  const int lane = tid & 31;
  const int n    = lane & 15;
  const int grp  = lane >> 4;
  const int m0   = blockIdx.x * 16;

  if (tid < 16) s_hsum[tid] = 0.0f;

  // ---------- Phase A: stats + inp(f16) tile + centered bb (wave w owns rows 4w..4w+3)
  for (int i = 0; i < 4; ++i) {
    const int row = 4*w + i;
    const int m   = m0 + row;
    const int b   = m / KDIM;
    const int k   = m - b*KDIM;
    const int idxb = x[b*7 + 2];
    const float* tp = theta + (size_t)m * TB;
    v4f tq[3], bq[3];
    float ts = 0.f, bs = 0.f;
#pragma unroll
    for (int j = 0; j < 3; ++j) {
      const int t = 4*lane + 128*j;          // 4-wide chunks; t<=332 implies all 4 valid
      v4f tv = {0.f,0.f,0.f,0.f}, bv = tv;
      if (t < TB) {
        tv = __builtin_nontemporal_load((const v4f*)(tp + t));  // streamed once
#pragma unroll
        for (int c = 0; c < 4; ++c) {
          int rr = idxb + 31 + t + c; if (rr >= PERIOD) rr -= PERIOD; // (idx+1-336+t)%366
          bv[c] = basis[rr*KDIM + k];
        }
      }
      tq[j] = tv; bq[j] = bv;
      ts += tv[0]+tv[1]+tv[2]+tv[3];
      bs += bv[0]+bv[1]+bv[2]+bv[3];
    }
#pragma unroll
    for (int off = 16; off > 0; off >>= 1) {
      ts += __shfl_xor(ts, off, 32);
      bs += __shfl_xor(bs, off, 32);
    }
    const float tmean = ts * (1.0f/TB);
    const float bmean = bs * (1.0f/TB);
#pragma unroll
    for (int j = 0; j < 3; ++j) {
      const int t = 4*lane + 128*j;
      if (t < TB) {
        const v4f tc = tq[j] - tmean;
        const v4f bc = bq[j] - bmean;
        v4h ih;
#pragma unroll
        for (int c = 0; c < 4; ++c) {
          const float d = tc[c] - bc[c];
          ih[c] = (_Float16)(1.0f - d*d / (tc[c]*tc[c] + 1e-6f));
        }
        *(v4h*)(&s_inp[row][t]) = ih;        // ds_store_b64
        *(v4f*)(&s_bb[row][t])  = bc;        // ds_store_b128
      }
    }
    if (lane < 16) s_inp[row][TB + lane] = (_Float16)0.0f;   // zero K padding
    if (lane == 0) { s_bbmean[row] = bmean; s_idx[row] = idxb; }
  }
  __syncthreads();

  // ---------- Phase B: z = tanh(inp @ W1^T + b1) via WMMA, fused h-partial = z @ W2
  // A fragments (this code's K order, same as B, not the ISA table order): lane holds M=lane&15; K contiguous grp*16..grp*16+15
  v16h afr[KK];
#pragma unroll
  for (int kk = 0; kk < KK; ++kk) {
    const _Float16* ap = &s_inp[n][kk*32 + grp*16];
    v8h alo = *(const v8h*)(ap);
    v8h ahi = *(const v8h*)(ap + 8);
    afr[kk] = __builtin_shufflevector(alo, ahi, 0,1,2,3,4,5,6,7,8,9,10,11,12,13,14,15);
  }

  float hpart[8];
#pragma unroll
  for (int j = 0; j < 8; ++j) hpart[j] = 0.f;

  for (int nt = w; nt < NT; nt += 4) {
    const int hrow = nt*16 + n;                    // hidden-unit index (B column)
    const int hc   = (hrow < HID) ? hrow : (HID-1); // clamped: no exec branches
    v8f acc = {};
#pragma unroll
    for (int kk = 0; kk < KK; ++kk) {
      // B fragment (32x16): lane-half holds 16 contiguous K values of column hrow
      const int t0  = kk*32 + grp*16;
      const int t0c = (kk == KK-1 && grp) ? (TB-16) : t0;  // only K-tail clamps
      const float* wp = W1 + hc*TB + t0c;
      v4f q0 = *(const v4f*)(wp);
      v4f q1 = *(const v4f*)(wp + 4);
      v4f q2 = *(const v4f*)(wp + 8);
      v4f q3 = *(const v4f*)(wp + 12);
      if (kk == KK-1) {                            // zero t>=336 (grp==1 lanes)
        const float zm = grp ? 0.0f : 1.0f;
        q0 *= zm; q1 *= zm; q2 *= zm; q3 *= zm;
      }
      union { v16h v; _Float16 e[16]; } bu;
#pragma unroll
      for (int c = 0; c < 4; ++c) {
        bu.e[c]      = (_Float16)q0[c];
        bu.e[4 + c]  = (_Float16)q1[c];
        bu.e[8 + c]  = (_Float16)q2[c];
        bu.e[12 + c] = (_Float16)q3[c];
      }
      acc = __builtin_amdgcn_wmma_f32_16x16x32_f16(false, afr[kk], false, bu.v,
                                                   (short)0, acc, false, false);
    }
    const float bia = B1v[hc];                      // garbage cols -> finite -> *0 below
    const float w2c = (hrow < HID) ? W2[hc] : 0.f;
#pragma unroll
    for (int j = 0; j < 8; ++j) {
      const float zin = acc[j] + bia;
      const float zt  = 1.0f - 2.0f/(__expf(2.0f*zin) + 1.0f);   // tanh
      hpart[j] += zt * w2c;
    }
  }
  // reduce over the 16 N-lanes of each half-wave; C/D layout: VGPR j -> M = j + 8*grp
#pragma unroll
  for (int j = 0; j < 8; ++j) {
    float v = hpart[j];
#pragma unroll
    for (int off = 8; off > 0; off >>= 1) v += __shfl_xor(v, off, 32);
    if (n == 0) atomicAdd(&s_hsum[j + 8*grp], v);
  }
  __syncthreads();

  // ---------- Phase C1: gate head, sample, KL partials
  if (tid < 16) {
    const int m = m0 + tid;
    const int k = m % KDIM;
    const float hs    = s_hsum[tid] + B2v[0];
    const float h     = 1.0f/(1.0f + __expf(-hs));
    const float h_new = __logf(h + 1e-8f/(1.0f - h + 1e-8f));
    const float alpha = 0.5f*h_new + 0.5f*weight[k];
    const float smp   = 1.0f/(1.0f + __expf(-(alpha + noise[m])*20.0f)); // /TEMP(0.05)
    s_sample[tid]   = smp;
    s_red[tid]      = h * __logf((h + 1e-5f)*2.0f);
    s_red[16 + tid] = 1.0f - h;
    s_red[32 + tid] = __logf((1.0f - h + 1e-5f)*2.0f);
  }
  __syncthreads();
  if (tid == 0) {
    float s1 = 0.f, s2 = 0.f, s3 = 0.f;
#pragma unroll
    for (int i = 0; i < 16; ++i) { s1 += s_red[i]; s2 += s_red[16+i]; s3 += s_red[32+i]; }
    atomicAdd(&gacc[0], s1);
    atomicAdd(&gacc[1], s2);
    atomicAdd(&gacc[2], s3);
  }

  // ---------- Phase C2: stream outputs (b128 nontemporal stores)
  for (int e = tid; e < 16*(TB/4); e += 128) {     // 16 rows x 84 quads
    const int row = e / (TB/4);
    const int t   = 4*(e - row*(TB/4));
    const v4f bb4 = *(const v4f*)(&s_bb[row][t]);
    __builtin_nontemporal_store(bb4 * s_sample[row],
                                (v4f*)(out + (size_t)(m0 + row)*TB + t));
  }
  const size_t FOFF = (size_t)ROWS * TB;
  for (int e = tid; e < 16*(TF/4); e += 128) {     // 16 rows x 24 quads
    const int row = e / (TF/4);
    const int t   = 4*(e - row*(TF/4));
    const int m   = m0 + row;
    const int k   = m % KDIM;
    const int ib  = s_idx[row];
    v4f bf;
#pragma unroll
    for (int c = 0; c < 4; ++c) {
      int rr = ib + 1 + t + c; if (rr >= PERIOD) rr -= PERIOD;
      bf[c] = basis[rr*KDIM + k] - s_bbmean[row];
    }
    __builtin_nontemporal_store(bf * s_sample[row],
                                (v4f*)(out + FOFF + (size_t)m*TF + t));
  }
}

__global__ void kl_finalize_kernel(const float* __restrict__ gacc,
                                   float* __restrict__ kl_out)
{
  const float inv = 1.0f / (float)ROWS;
  kl_out[0] = gacc[0]*inv + (gacc[1]*inv) * (gacc[2]*inv);
}

extern "C" void kernel_launch(void* const* d_in, const int* in_sizes, int n_in,
                              void* d_out, int out_size, void* d_ws, size_t ws_size,
                              hipStream_t stream) {
  const float* theta  = (const float*)d_in[0];
  const int*   x      = (const int*)  d_in[1];
  const float* basis  = (const float*)d_in[2];
  const float* W1     = (const float*)d_in[3];
  const float* b1     = (const float*)d_in[4];
  const float* W2     = (const float*)d_in[5];
  const float* b2     = (const float*)d_in[6];
  const float* weight = (const float*)d_in[7];
  const float* noise  = (const float*)d_in[8];
  float* out  = (float*)d_out;
  float* gacc = (float*)d_ws;

  hipMemsetAsync(gacc, 0, 3 * sizeof(float), stream);   // graph-capturable
  nbeats_gate_kernel<<<ROWS/16, 128, 0, stream>>>(theta, x, basis, W1, b1, W2, b2,
                                                  weight, noise, out, gacc);
  kl_finalize_kernel<<<1, 1, 0, stream>>>(gacc, out + (out_size - 1));
}
